// tf_Cross_Attention_20349555049000
// MI455X (gfx1250) — compile-verified, matrix operand order checked by reading, not run on hardware
//
#include <hip/hip_runtime.h>
#include <hip/hip_bf16.h>

typedef _Float16 half_t;
typedef __attribute__((ext_vector_type(2)))  _Float16 v2h;
typedef __attribute__((ext_vector_type(8)))  _Float16 v8h;
typedef __attribute__((ext_vector_type(16))) _Float16 v16h;
typedef __attribute__((ext_vector_type(8)))  float    v8f;

#define CCH   128
#define HH    192
#define WW    192
#define HWSZ  36864        // H*W
#define MTOT  294912L      // N*H*W
#define EPSB  1e-5f

// padded LDS row strides (halves): conflict-free 16-lane fragment loads
#define LDA   136          // 128-wide tiles (68 words -> banks 4r+kw)
#define LDP   200          // 192-wide P tile
#define LDG   72           // 64-wide gT tile

// ---------------------------------------------------------------------------
// CDNA5 16-bit WMMA fragment: per lane, row r = lane&15, K sub-block selected
// by half-wave (kb = 0/8, plus +16). Two contiguous 16B runs per lane ->
// two ds_load_b128. Serves A directly and B from transposed [n][k] tiles.
// ---------------------------------------------------------------------------
__device__ __forceinline__ v16h load_frag_row(const half_t* base, int ld, int r0, int k0) {
  int lane = threadIdx.x & 31;
  int r  = lane & 15;
  int kb = (lane >> 4) << 3;            // 0 or 8
  const half_t* p = base + (r0 + r) * ld + k0 + kb;
  union { v16h v; v8h h[2]; } u;
  u.h[0] = *(const v8h*)(p);
  u.h[1] = *(const v8h*)(p + 16);
  return u.v;
}

__device__ __forceinline__ v8f wmma_f16(v16h a, v16h b, v8f c) {
  return __builtin_amdgcn_wmma_f32_16x16x32_f16(false, a, false, b, (short)0, c, false, false);
}

// ===========================================================================
// Kernel 0a: pre-transpose weights to [n][k] f16. Tiles 0..6 = W7,
// tiles 7..8 = Wa halves. 288KB total, one-shot.
// ===========================================================================
__global__ void __launch_bounds__(256)
k0_transpose_w(const float* __restrict__ W7, const float* __restrict__ Wa,
               half_t* __restrict__ wT)
{
  long idx = (long)blockIdx.x * 256 + threadIdx.x;   // 9*16384 elements
  int tile = (int)(idx >> 14);
  int e    = (int)(idx & 16383);
  int n = e >> 7, k = e & 127;
  float v = (tile < 7) ? W7[(long)tile * 16384 + k * 128 + n]
                       : Wa[((long)(tile - 7) * 128 + k) * 128 + n];
  wT[idx] = (half_t)v;
}

// Kernel 0b: fold BN into per-channel (scale, bias). Rows 0..6 = convs,
// row 7 = final BN (ma/va/ba).
__global__ void __launch_bounds__(256)
k0_bn_tab(const float* __restrict__ m7, const float* __restrict__ v7,
          const float* __restrict__ b7, const float* __restrict__ ma,
          const float* __restrict__ va, const float* __restrict__ ba,
          float2* __restrict__ bnTab)
{
  int idx = blockIdx.x * 256 + threadIdx.x;          // 8*128
  if (idx >= 1024) return;
  int wi = idx >> 7, ch = idx & 127;
  float mm, vv, bb;
  if (wi < 7) { mm = m7[wi * 128 + ch]; vv = v7[wi * 128 + ch]; bb = b7[wi * 128 + ch]; }
  else        { mm = ma[ch]; vv = va[ch]; bb = ba[ch]; }
  float sc = rsqrtf(vv + EPSB);
  bnTab[idx] = make_float2(sc, bb - mm * sc);
}

// ===========================================================================
// Kernel 1: theta/phi/g (+ReLU) and the two skip branches (W7[4], W7[6]),
// fused conv1x1 + BN, f16 pixel-major [p][c] outputs.
// 256 threads = 8 waves; tile = 256 pixels x 128 ch; wave = 32x128 slab
// (2 A-frags share each B-frag: 1.25 ds_load_b128 per WMMA). ~104KB LDS.
// ===========================================================================
__global__ void __launch_bounds__(256)
k1_qkv(const float* __restrict__ x, const half_t* __restrict__ wT,
       const float2* __restrict__ bnTab,
       half_t* __restrict__ th, half_t* __restrict__ ph, half_t* __restrict__ gq,
       half_t* __restrict__ fhsc, half_t* __restrict__ fvsc)
{
  extern __shared__ __align__(16) char smem[];
  half_t* bufx = (half_t*)smem;                    // [256][LDA] activations [m][k]
  half_t* bufw = (half_t*)(smem + 256 * LDA * 2);  // [128][LDA] weights [n][k] / C staging

  const int tid = threadIdx.x, wq = tid >> 5, lane = tid & 31;
  const int nlo = lane & 15, hi = lane >> 4;
  const long pbase = (long)blockIdx.x * 256;
  const int  nimg  = (int)(pbase / HWSZ);
  const long off   = pbase - (long)nimg * HWSZ;

  // stage X tile: float4 loads (coalesced along pixels), packed v2h LDS stores
#pragma unroll
  for (int t = 0; t < 16; ++t) {
    int e  = tid + t * 256;            // 4096 groups: 64 ch-pairs x 64 m-quads
    int cp = e >> 6, mg = e & 63;
    const float* s0 = x + ((long)(nimg * CCH + 2 * cp)) * HWSZ + off + mg * 4;
    float4 a0 = *(const float4*)s0;
    float4 a1 = *(const float4*)(s0 + HWSZ);
    half_t* d = bufx + (mg * 4) * LDA + 2 * cp;
    *(v2h*)(d)           = v2h{(half_t)a0.x, (half_t)a1.x};
    *(v2h*)(d + LDA)     = v2h{(half_t)a0.y, (half_t)a1.y};
    *(v2h*)(d + 2*LDA)   = v2h{(half_t)a0.z, (half_t)a1.z};
    *(v2h*)(d + 3*LDA)   = v2h{(half_t)a0.w, (half_t)a1.w};
  }

  const int wsel[5] = {0, 1, 2, 4, 6};
  half_t* const outs[5] = {th, ph, gq, fhsc, fvsc};

  for (int br = 0; br < 5; ++br) {
    const int wi = wsel[br];
    __syncthreads();                               // bufw free
    {
      const half_t* src = wT + (long)wi * 16384;
#pragma unroll
      for (int t = 0; t < 8; ++t) {
        int e = tid + t * 256;                     // 2048 v8h
        int n = e >> 4, kv = (e & 15) << 3;
        *(v8h*)(bufw + n * LDA + kv) = *(const v8h*)(src + n * 128 + kv);
      }
    }
    __syncthreads();

    v8f acc[16];
    { v8f z = {};
#pragma unroll
      for (int i = 0; i < 16; ++i) acc[i] = z; }

#pragma unroll
    for (int k = 0; k < 4; ++k) {
      v16h a0 = load_frag_row(bufx, LDA, wq * 32,      k * 32);
      v16h a1 = load_frag_row(bufx, LDA, wq * 32 + 16, k * 32);
#pragma unroll
      for (int nt = 0; nt < 8; ++nt) {
        v16h b = load_frag_row(bufw, LDA, nt * 16, k * 32);
        acc[nt]     = wmma_f16(a0, b, acc[nt]);
        acc[8 + nt] = wmma_f16(a1, b, acc[8 + nt]);
      }
    }
    __syncthreads();                               // GEMM reads of bufw done

    // BN affine (+ReLU for theta/phi/g) -> staged + coalesced stores, 2 halves
    half_t* ob = outs[br];
    for (int h = 0; h < 2; ++h) {
      if ((wq >> 2) == h) {                        // waves owning rows [128h,128h+128)
#pragma unroll
        for (int nt = 0; nt < 8; ++nt) {
          int ch = nt * 16 + nlo;
          float2 sb = bnTab[wi * 128 + ch];
#pragma unroll
          for (int mt = 0; mt < 2; ++mt)
#pragma unroll
            for (int r = 0; r < 8; ++r) {
              int m = wq * 32 + mt * 16 + r + hi * 8;
              float y = acc[mt * 8 + nt][r] * sb.x + sb.y;
              if (br < 3) y = fmaxf(y, 0.f);
              bufw[(m - h * 128) * LDA + ch] = (half_t)y;
            }
        }
      }
      __syncthreads();
#pragma unroll
      for (int t = 0; t < 8; ++t) {
        int e = tid + t * 256;                     // 2048 v8h per half
        int m = e >> 4, cv = (e & 15) << 3;
        *(v8h*)(ob + (pbase + h * 128 + m) * 128 + cv) =
            *(const v8h*)(bufw + m * LDA + cv);
      }
      __syncthreads();
    }
  }
}

// ===========================================================================
// Kernel 2: fused axial attention for one (sequence, 64-query block).
// mode 0: rows (pixel stride 1); mode 1: columns (stride W).
// 128 threads = 4 waves, one 16-query tile per wave. Scores never hit HBM.
// ===========================================================================
#define K2_RA_BYTES  (128 * LDG * 2)     // 18432: max(th tile 64*LDA*2, gT 128*LDG*2)
#define K2_RB_BYTES  (96 * LDA * 2)      // 26112: max(ph 96*LDA*2, P 64*LDP*2)

__global__ void __launch_bounds__(128)
k2_attn(const half_t* __restrict__ th, const half_t* __restrict__ ph,
        const half_t* __restrict__ gq, half_t* __restrict__ outb, int mode)
{
  __shared__ __align__(16) char smem[K2_RA_BYTES + K2_RB_BYTES];
  half_t* regA = (half_t*)smem;                  // th tile -> gT chunk -> out staging
  half_t* regB = (half_t*)(smem + K2_RA_BYTES);  // ph chunk -> P

  const int tid = threadIdx.x, wq = tid >> 5, lane = tid & 31;
  const int nlo = lane & 15, hi = lane >> 4;
  const int b = blockIdx.y, qb = blockIdx.x;
  long p0; int pstr;
  if (mode == 0) { p0 = (long)b * WW; pstr = 1; }
  else           { int n = b / WW, w = b - n * WW; p0 = (long)n * HWSZ + w; pstr = WW; }
  const int q0 = qb * 64;

  // stage th[q0..q0+63][0..127] : v8h global -> v8h LDS (padded rows)
#pragma unroll
  for (int t = 0; t < 8; ++t) {
    int e = tid + t * 128;                   // 1024 vectors
    int m = e >> 4, cv = (e & 15) << 3;
    *(v8h*)(regA + m * LDA + cv) =
        *(const v8h*)(th + (p0 + (long)(q0 + m) * pstr) * 128 + cv);
  }
  __syncthreads();

  v16h afr[4];
#pragma unroll
  for (int k = 0; k < 4; ++k) afr[k] = load_frag_row(regA, LDA, wq * 16, k * 32);

  v8f sacc[12];
  { v8f z = {};
#pragma unroll
    for (int i = 0; i < 12; ++i) sacc[i] = z; }

  // S = th . ph^T in two 96-key chunks (ph^T fragment == A layout on [j][c])
  for (int cc = 0; cc < 2; ++cc) {
    __syncthreads();
#pragma unroll
    for (int t = 0; t < 12; ++t) {
      int e = tid + t * 128;                 // 1536 vectors
      int j = e >> 4, cv = (e & 15) << 3;
      *(v8h*)(regB + j * LDA + cv) =
          *(const v8h*)(ph + (p0 + (long)(cc * 96 + j) * pstr) * 128 + cv);
    }
    __syncthreads();
#pragma unroll
    for (int k = 0; k < 4; ++k)
#pragma unroll
      for (int jt = 0; jt < 6; ++jt) {
        v16h bfr = load_frag_row(regB, LDA, jt * 16, k * 32);
        sacc[cc * 6 + jt] = wmma_f16(afr[k], bfr, sacc[cc * 6 + jt]);
      }
  }

  // softmax over j (192): 16-lane shuffle reductions match the C-layout rows
  const float scl = 0.08838834764831845f;    // 1/sqrt(128)
  float inv[8];
#pragma unroll
  for (int r = 0; r < 8; ++r) {
    float mx = -3.4e38f;
#pragma unroll
    for (int t = 0; t < 12; ++t) mx = fmaxf(mx, sacc[t][r]);
#pragma unroll
    for (int o = 1; o < 16; o <<= 1) mx = fmaxf(mx, __shfl_xor(mx, o, 32));
    float s = 0.f;
#pragma unroll
    for (int t = 0; t < 12; ++t) {
      float e = __expf((sacc[t][r] - mx) * scl);
      sacc[t][r] = e; s += e;
    }
#pragma unroll
    for (int o = 1; o < 16; o <<= 1) s += __shfl_xor(s, o, 32);
    inv[r] = 1.f / s;
  }

  // P (f16) into LDS (overlays ph chunk region, padded stride)
  half_t* P = regB;
  __syncthreads();
#pragma unroll
  for (int t = 0; t < 12; ++t)
#pragma unroll
    for (int r = 0; r < 8; ++r) {
      int m = r + hi * 8;
      P[(wq * 16 + m) * LDP + t * 16 + nlo] = (half_t)(sacc[t][r] * inv[r]);
    }

  // out = P @ g; g streamed in 64-key chunks, transposed to [c][j] in LDS so
  // B-fragments are two ds_load_b128 per lane.
  half_t* gT = regA;
  v8f oacc[8];
  { v8f z = {};
#pragma unroll
    for (int i = 0; i < 8; ++i) oacc[i] = z; }
  for (int gc = 0; gc < 3; ++gc) {
    __syncthreads();
#pragma unroll
    for (int jq = 0; jq < 8; ++jq) {         // thread = channel; coalesced u16
      v8h vv;
#pragma unroll
      for (int i = 0; i < 8; ++i)
        vv[i] = gq[(p0 + (long)(gc * 64 + jq * 8 + i) * pstr) * 128 + tid];
      *(v8h*)(gT + tid * LDG + jq * 8) = vv;
    }
    __syncthreads();
#pragma unroll
    for (int ks = 0; ks < 2; ++ks) {
      v16h a = load_frag_row(P, LDP, wq * 16, gc * 64 + ks * 32);
#pragma unroll
      for (int nt = 0; nt < 8; ++nt) {
        v16h bfr = load_frag_row(gT, LDG, nt * 16, ks * 32);
        oacc[nt] = wmma_f16(a, bfr, oacc[nt]);
      }
    }
  }

  // stage out tile [q][ch] in LDS, then coalesced v8h global stores
  half_t* ostg = regA;
  __syncthreads();                           // gT reads done
#pragma unroll
  for (int nt = 0; nt < 8; ++nt) {
    int ch = nt * 16 + nlo;
#pragma unroll
    for (int r = 0; r < 8; ++r) {
      int m = wq * 16 + r + hi * 8;
      ostg[m * LDA + ch] = (half_t)oacc[nt][r];
    }
  }
  __syncthreads();
#pragma unroll
  for (int t = 0; t < 8; ++t) {
    int e = tid + t * 128;                   // 1024 vectors
    int m = e >> 4, cv = (e & 15) << 3;
    *(v8h*)(outb + (p0 + (long)(q0 + m) * pstr) * 128 + cv) =
        *(const v8h*)(ostg + m * LDA + cv);
  }
}

// ===========================================================================
// Kernel 3: fused  fh=BN(attnH@W7[3])+skip,ReLU ; fv likewise ; out=BN(cat@Wa)
// ReLU; transposed coalesced NCHW f32 store via padded LDS.
// 256-pixel tiles, wave = 32x128 slab. ~104KB dynamic LDS.
// ===========================================================================
__global__ void __launch_bounds__(256)
k3_final(const half_t* __restrict__ aH, const half_t* __restrict__ aV,
         const half_t* __restrict__ fhsc, const half_t* __restrict__ fvsc,
         const half_t* __restrict__ wT, const float2* __restrict__ bnTab,
         float* __restrict__ out)
{
  extern __shared__ __align__(16) char smem[];
  half_t* bufx = (half_t*)smem;                    // [256][LDA] activation [m][k]
  half_t* bufw = (half_t*)(smem + 256 * LDA * 2);  // [128][LDA] weight [n][k]
  float*  tr   = (float*)smem;                     // 64*129 f32 transpose (reused)

  const int tid = threadIdx.x, wq = tid >> 5, lane = tid & 31;
  const int nlo = lane & 15, hi = lane >> 4;
  const long pbase = (long)blockIdx.x * 256;
  const int  nimg  = (int)(pbase / HWSZ);
  const long off   = pbase - (long)nimg * HWSZ;

  v8f acc2[16];
  { v8f z = {};
#pragma unroll
    for (int i = 0; i < 16; ++i) acc2[i] = z; }

  for (int pass = 0; pass < 2; ++pass) {
    const half_t* ain  = pass ? aV : aH;
    const half_t* scin = pass ? fvsc : fhsc;
    const int wi = pass ? 5 : 3;

    __syncthreads();
#pragma unroll
    for (int t = 0; t < 16; ++t) {           // 4096 v8h activation copy
      int e = tid + t * 256;
      int m = e >> 4, cv = (e & 15) << 3;
      *(v8h*)(bufx + m * LDA + cv) = *(const v8h*)(ain + (pbase + m) * 128 + cv);
    }
    {
      const half_t* src = wT + (long)wi * 16384;
#pragma unroll
      for (int t = 0; t < 8; ++t) {
        int e = tid + t * 256;
        int n = e >> 4, kv = (e & 15) << 3;
        *(v8h*)(bufw + n * LDA + kv) = *(const v8h*)(src + n * 128 + kv);
      }
    }
    __syncthreads();

    v8f acc[16];
    { v8f z = {};
#pragma unroll
      for (int i = 0; i < 16; ++i) acc[i] = z; }
#pragma unroll
    for (int k = 0; k < 4; ++k) {
      v16h a0 = load_frag_row(bufx, LDA, wq * 32,      k * 32);
      v16h a1 = load_frag_row(bufx, LDA, wq * 32 + 16, k * 32);
#pragma unroll
      for (int nt = 0; nt < 8; ++nt) {
        v16h b = load_frag_row(bufw, LDA, nt * 16, k * 32);
        acc[nt]     = wmma_f16(a0, b, acc[nt]);
        acc[8 + nt] = wmma_f16(a1, b, acc[8 + nt]);
      }
    }
    __syncthreads();   // all bufx/bufw reads done

    // BN + skip + ReLU -> f16 back into bufx ; stage Wa half (transposed)
#pragma unroll
    for (int nt = 0; nt < 8; ++nt) {
      int ch = nt * 16 + nlo;
      float2 sb = bnTab[wi * 128 + ch];
#pragma unroll
      for (int mt = 0; mt < 2; ++mt)
#pragma unroll
        for (int r = 0; r < 8; ++r) {
          int m = wq * 32 + mt * 16 + r + hi * 8;
          float y = acc[mt * 8 + nt][r] * sb.x + sb.y
                    + (float)scin[(pbase + m) * 128 + ch];
          bufx[m * LDA + ch] = (half_t)fmaxf(y, 0.f);
        }
    }
    {
      const half_t* src = wT + (long)(7 + pass) * 16384;
#pragma unroll
      for (int t = 0; t < 8; ++t) {
        int e = tid + t * 256;
        int n = e >> 4, kv = (e & 15) << 3;
        *(v8h*)(bufw + n * LDA + kv) = *(const v8h*)(src + n * 128 + kv);
      }
    }
    __syncthreads();
#pragma unroll
    for (int k = 0; k < 4; ++k) {
      v16h a0 = load_frag_row(bufx, LDA, wq * 32,      k * 32);
      v16h a1 = load_frag_row(bufx, LDA, wq * 32 + 16, k * 32);
#pragma unroll
      for (int nt = 0; nt < 8; ++nt) {
        v16h b = load_frag_row(bufw, LDA, nt * 16, k * 32);
        acc2[nt]     = wmma_f16(a0, b, acc2[nt]);
        acc2[8 + nt] = wmma_f16(a1, b, acc2[8 + nt]);
      }
    }
  }

  // final BN + ReLU, transposed store (four 64-row quarters, pad 129 -> no
  // bank conflicts, coalesced 4B stores along pixel dim of NCHW output)
  for (int h = 0; h < 4; ++h) {
    __syncthreads();
    if ((wq >> 1) == h) {                    // waves owning rows [64h, 64h+64)
#pragma unroll
      for (int nt = 0; nt < 8; ++nt) {
        int ch = nt * 16 + nlo;
        float2 sb = bnTab[7 * 128 + ch];
#pragma unroll
        for (int mt = 0; mt < 2; ++mt)
#pragma unroll
          for (int r = 0; r < 8; ++r) {
            int m  = wq * 32 + mt * 16 + r + hi * 8;
            int lr = m - h * 64;
            tr[lr * 129 + ch] = fmaxf(acc2[mt * 8 + nt][r] * sb.x + sb.y, 0.f);
          }
      }
    }
    __syncthreads();
#pragma unroll
    for (int t = 0; t < 32; ++t) {
      int e  = tid + t * 256;
      int ch = e >> 6, mm = e & 63;
      out[((long)(nimg * CCH + ch)) * HWSZ + off + h * 64 + mm] = tr[mm * 129 + ch];
    }
  }
}

// ===========================================================================
extern "C" void kernel_launch(void* const* d_in, const int* in_sizes, int n_in,
                              void* d_out, int out_size, void* d_ws, size_t ws_size,
                              hipStream_t stream) {
  const float* f  = (const float*)d_in[0];
  const float* W7 = (const float*)d_in[1];
  const float* m7 = (const float*)d_in[2];
  const float* v7 = (const float*)d_in[3];
  const float* b7 = (const float*)d_in[4];
  const float* Wa = (const float*)d_in[5];
  const float* ma = (const float*)d_in[6];
  const float* va = (const float*)d_in[7];
  const float* ba = (const float*)d_in[8];
  float* out = (float*)d_out;

  const long ME = MTOT * CCH;                 // elements per f16 tensor
  half_t* ws   = (half_t*)d_ws;
  half_t* th   = ws;
  half_t* ph   = ws + 1 * ME;
  half_t* gq   = ws + 2 * ME;
  half_t* fhsc = ws + 3 * ME;
  half_t* fvsc = ws + 4 * ME;
  half_t* aH   = ws + 5 * ME;
  half_t* aV   = ws + 6 * ME;
  half_t* wT   = ws + 7 * ME;                 // 9 * 16384 transposed weights
  float2* bnT  = (float2*)(ws + 7 * ME + 9 * 16384);  // 8 * 128 (sc, bias)

  const size_t ldsK13 = (256u * LDA + 128u * LDA) * 2;  // ~104KB dynamic LDS

  // 0) one-shot setup: transposed weights + folded BN coefficients
  k0_transpose_w<<<dim3(576), 256, 0, stream>>>(W7, Wa, wT);
  k0_bn_tab<<<dim3(4), 256, 0, stream>>>(m7, v7, b7, ma, va, ba, bnT);

  // 1) theta/phi/g + skip branches (256-pixel tiles)
  k1_qkv<<<dim3(1152), 256, ldsK13, stream>>>(f, wT, bnT, th, ph, gq, fhsc, fvsc);

  // 2) row attention (1536 rows x 3 query blocks), then column attention
  k2_attn<<<dim3(3, 1536), 128, 0, stream>>>(th, ph, gq, aH, 0);
  k2_attn<<<dim3(3, 1536), 128, 0, stream>>>(th, ph, gq, aV, 1);

  // 3) fused output stage (256-pixel tiles)
  k3_final<<<dim3(1152), 256, ldsK13, stream>>>(aH, aV, fhsc, fvsc, wT, bnT, out);
}
